// FVN_41764261986386
// MI455X (gfx1250) — matrix operand order repaired and checked by reading, not compiled, not run on hardware
//
#include <hip/hip_runtime.h>

// ---------------------------------------------------------------------------
// FVN reverb: out[b,n] = sum_m velvet[b,m]*gains[b,m/300]*sap[n-m] + fir (n<4096)
// Banded-Toeplitz GEMM on v_wmma_f32_16x16x32_f16.
// Per wave: 16(batch) x 64(n) tile = 4 accumulators sharing one A fragment.
// Software-pipelined K loop: next iteration's A/B loads issued before this
// iteration's 4 WMMAs so load latency is hidden behind matrix ops.
// ---------------------------------------------------------------------------

typedef _Float16 h16 __attribute__((ext_vector_type(16)));
typedef _Float16 h8  __attribute__((ext_vector_type(8)));
typedef float    f8  __attribute__((ext_vector_type(8)));

#define SEG_NUM    160
#define IR_LENGTH  48000
#define SAP_LEN    9000
#define BATCH      64
#define FIR_LEN    4096
#define STRIDE     300                       // IR_LENGTH / SEG_NUM
#define OUT_LEN    56999                     // 300*159 + (300+9000-1)
#define SAP_PAD    96                        // zero pad absorbs band edges (4 subtiles)
#define SAP_BUF    9200                      // 96 + 9000 + 104
#define N_SUPER    ((OUT_LEN + 63) / 64)     // 891 super-tiles of 64 n
#define B_TILES    (BATCH / 16)              // 4

// --- prep 1: xg[b,m] = (f16)(velvet[b,m] * gains[b, m/300]) ----------------
__global__ void prep_xg(const float* __restrict__ velvet,
                        const float* __restrict__ gains,
                        _Float16* __restrict__ xg) {
    int t = blockIdx.x * blockDim.x + threadIdx.x;
    if (t >= BATCH * IR_LENGTH) return;
    int b = t / IR_LENGTH;
    int m = t - b * IR_LENGTH;
    xg[t] = (_Float16)(velvet[t] * gains[b * SEG_NUM + m / STRIDE]);
}

// --- prep 2: reversed, zero-padded sap in f16 ------------------------------
__global__ void prep_sap(const float* __restrict__ sap,
                         _Float16* __restrict__ sapr) {
    int i = blockIdx.x * blockDim.x + threadIdx.x;
    if (i >= SAP_BUF) return;
    int j = i - SAP_PAD;
    sapr[i] = (j >= 0 && j < SAP_LEN) ? (_Float16)sap[(SAP_LEN - 1) - j]
                                      : (_Float16)0.0f;
}

#define WMMA_F16(A, Bf, C)                                              \
    __builtin_amdgcn_wmma_f32_16x16x32_f16(false, (A), false, (Bf),     \
                                           (short)0, (C), false, false)

union AF { h16 v; h8 p[2]; };

// --- main conv: one wave computes a 16(batch) x 64(n) output tile ----------
__global__ void __launch_bounds__(256)
fvn_conv(const _Float16* __restrict__ xg,
         const _Float16* __restrict__ sapr,
         float* __restrict__ out) {
    const int lane = threadIdx.x & 31;
    const int wave = threadIdx.x >> 5;
    const int tile = blockIdx.x * 8 + wave;
    if (tile >= N_SUPER * B_TILES) return;

    const int ntile = tile >> 2;           // 64-wide n super-tile
    const int btile = tile & 3;            // 16-wide batch tile
    const int n0 = ntile * 64;
    const int b0 = btile * 16;

    const int row = lane & 15;             // A: M row / B,D: N col
    const int hi  = lane >> 4;
    const int kbA = hi * 16;               // A K-group base (contiguous k = 16*hi + s)
    const int kbB = hi * 16;               // B K-group base (contiguous k = 16*hi + s, as A)

    // m-range contributing to n in [n0, n0+64): m in [n0-8999, n0+63], 32-aligned
    int m_lo = n0 - (SAP_LEN - 1);
    if (m_lo < 0) m_lo = 0;
    m_lo &= ~31;
    int m_hi = n0 + 64;
    if (m_hi > IR_LENGTH) m_hi = IR_LENGTH;
    m_hi = (m_hi + 31) & ~31;              // IR_LENGTH is a multiple of 32

    // A: row-major xg, two aligned 16B loads per lane per K-step (shared x4)
    const _Float16* paBase = xg + (size_t)(b0 + row) * IR_LENGTH + kbA;
    // B: Toeplitz band from reversed sap; subtile j is shifted by -16j halves
    const _Float16* pbBase = sapr + (SAP_PAD + (SAP_LEN - 1) - n0 + kbB - row);

    f8 acc0 = {}, acc1 = {}, acc2 = {}, acc3 = {};

    // ---- software pipeline: prologue load (m-range is never empty) --------
    AF  aC;
    h16 b0C, b1C, b2C, b3C;
    {
        const _Float16* pa = paBase + m_lo;
        const _Float16* pb = pbBase + m_lo;
        aC.p[0] = *(const h8*)(pa);
        aC.p[1] = *(const h8*)(pa + 8);
        __builtin_memcpy(&b0C, pb,      sizeof(h16));
        __builtin_memcpy(&b1C, pb - 16, sizeof(h16));
        __builtin_memcpy(&b2C, pb - 32, sizeof(h16));
        __builtin_memcpy(&b3C, pb - 48, sizeof(h16));
    }

    // ---- steady state: issue next loads, then 4 WMMAs on current ----------
    for (int m0 = m_lo; m0 + 32 < m_hi; m0 += 32) {
        AF  aN;
        h16 b0N, b1N, b2N, b3N;
        const _Float16* pa = paBase + m0 + 32;
        const _Float16* pb = pbBase + m0 + 32;
        aN.p[0] = *(const h8*)(pa);
        aN.p[1] = *(const h8*)(pa + 8);
        __builtin_memcpy(&b0N, pb,      sizeof(h16));
        __builtin_memcpy(&b1N, pb - 16, sizeof(h16));
        __builtin_memcpy(&b2N, pb - 32, sizeof(h16));
        __builtin_memcpy(&b3N, pb - 48, sizeof(h16));

        acc0 = WMMA_F16(aC.v, b0C, acc0);
        acc1 = WMMA_F16(aC.v, b1C, acc1);
        acc2 = WMMA_F16(aC.v, b2C, acc2);
        acc3 = WMMA_F16(aC.v, b3C, acc3);

        aC = aN;
        b0C = b0N; b1C = b1N; b2C = b2N; b3C = b3N;
    }

    // ---- epilogue: last K-step ---------------------------------------------
    acc0 = WMMA_F16(aC.v, b0C, acc0);
    acc1 = WMMA_F16(aC.v, b1C, acc1);
    acc2 = WMMA_F16(aC.v, b2C, acc2);
    acc3 = WMMA_F16(aC.v, b3C, acc3);

    // D layout: VGPR v, lane -> M = v + 8*hi (batch), N = lane&15 (n)
    const size_t rowBase = (size_t)(b0 + hi * 8) * OUT_LEN;
    {
        int n = n0 + row;
        if (n < OUT_LEN) {
            #pragma unroll
            for (int v = 0; v < 8; ++v)
                out[rowBase + (size_t)v * OUT_LEN + n] = acc0[v];
        }
        n += 16;
        if (n < OUT_LEN) {
            #pragma unroll
            for (int v = 0; v < 8; ++v)
                out[rowBase + (size_t)v * OUT_LEN + n] = acc1[v];
        }
        n += 16;
        if (n < OUT_LEN) {
            #pragma unroll
            for (int v = 0; v < 8; ++v)
                out[rowBase + (size_t)v * OUT_LEN + n] = acc2[v];
        }
        n += 16;
        if (n < OUT_LEN) {
            #pragma unroll
            for (int v = 0; v < 8; ++v)
                out[rowBase + (size_t)v * OUT_LEN + n] = acc3[v];
        }
    }
}

// --- epilogue: add FIR head ------------------------------------------------
__global__ void add_fir(const float* __restrict__ fir, float* __restrict__ out) {
    int t = blockIdx.x * blockDim.x + threadIdx.x;
    if (t >= BATCH * FIR_LEN) return;
    int b = t / FIR_LEN;
    int n = t - b * FIR_LEN;
    out[(size_t)b * OUT_LEN + n] += fir[t];
}

extern "C" void kernel_launch(void* const* d_in, const int* in_sizes, int n_in,
                              void* d_out, int out_size, void* d_ws, size_t ws_size,
                              hipStream_t stream) {
    const float* gains  = (const float*)d_in[0];   // (64, 160)
    const float* fir    = (const float*)d_in[1];   // (64, 4096)
    const float* velvet = (const float*)d_in[2];   // (64, 48000)
    const float* sap    = (const float*)d_in[3];   // (9000,)
    float* out = (float*)d_out;                    // (64, 56999)

    _Float16* xg   = (_Float16*)d_ws;                                  // 6.14 MB
    _Float16* sapr = (_Float16*)((char*)d_ws +
                                 (size_t)BATCH * IR_LENGTH * sizeof(_Float16));

    prep_xg <<<(BATCH * IR_LENGTH + 255) / 256, 256, 0, stream>>>(velvet, gains, xg);
    prep_sap<<<(SAP_BUF + 255) / 256,          256, 0, stream>>>(sap, sapr);

    const int tiles = N_SUPER * B_TILES;           // 3564 wave-tiles
    fvn_conv<<<(tiles + 7) / 8, 256, 0, stream>>>(xg, sapr, out);

    add_fir <<<(BATCH * FIR_LEN + 255) / 256, 256, 0, stream>>>(fir, out);
}
